// CoConvBlock_44032004718972
// MI455X (gfx1250) — compile-verified, matrix operand order checked by reading, not run on hardware
//
#include <hip/hip_runtime.h>

#define B_    16
#define CI_   200
#define CO_   128
#define HH_   128
#define WW_   128
#define HW_   (HH_*WW_)
#define NN_   1024
#define CPAD_ 224

typedef __attribute__((ext_vector_type(16))) __bf16 v16bf;
typedef __attribute__((ext_vector_type(8)))  float  v8f;
typedef __attribute__((ext_vector_type(4)))  int    v4i;

union FragAB { v16bf v; unsigned int u[8]; uint4 q[2]; };
union FragC  { v8f   v; float        f[8]; };

static __device__ __forceinline__ unsigned short f2bf(float f) {
    unsigned int u = __float_as_uint(f);
    u += 0x7fffu + ((u >> 16) & 1u);        // round-to-nearest-even
    return (unsigned short)(u >> 16);
}
static __device__ __forceinline__ float sigmoidf_(float x) {
    return 1.0f / (1.0f + __expf(-x));
}

// ---------------------------------------------------------------------------
// Async global->LDS copy (CDNA5 GLOBAL_LOAD_ASYNC_TO_LDS_B128, ASYNCcnt),
// with a synchronous fallback if the toolchain lacks the builtin.
// Builtin signature (from clang diagnostic): (v4i AS1*, v4i AS3*, Imm, Imm).
// ---------------------------------------------------------------------------
#if __has_builtin(__builtin_amdgcn_global_load_async_to_lds_b128)
#define USE_ASYNC_LDS 1
typedef __attribute__((address_space(1))) v4i* gv4i_ptr;
typedef __attribute__((address_space(3))) v4i* lv4i_ptr;
#else
#define USE_ASYNC_LDS 0
#endif

static __device__ __forceinline__ void cp16_lds(unsigned short* dst,
                                                const unsigned short* src) {
#if USE_ASYNC_LDS
    __builtin_amdgcn_global_load_async_to_lds_b128(
        (gv4i_ptr)(v4i*)src, (lv4i_ptr)(v4i*)dst, 0, 0);
#else
    *(uint4*)dst = *(const uint4*)src;
#endif
}

static __device__ __forceinline__ void async_wait_all() {
#if USE_ASYNC_LDS
#if __has_builtin(__builtin_amdgcn_s_wait_asynccnt)
    __builtin_amdgcn_s_wait_asynccnt(0);
#else
    asm volatile("s_wait_asynccnt 0" ::: "memory");
#endif
#endif
}

// ---------------------------------------------------------------------------
// Fold BN into weights (bf16) and biases; zero the X_mean accumulator.
// ---------------------------------------------------------------------------
__global__ __launch_bounds__(256) void prep_params_kernel(
    const float* __restrict__ bn1_g, const float* __restrict__ bn1_b,
    const float* __restrict__ bn1_m, const float* __restrict__ bn1_v,
    const float* __restrict__ bn2_g, const float* __restrict__ bn2_b,
    const float* __restrict__ bn2_m, const float* __restrict__ bn2_v,
    const float* __restrict__ W_lin, const float* __restrict__ b_lin,
    const float* __restrict__ W_sim,
    const float* __restrict__ W_final, const float* __restrict__ b_final,
    unsigned short* __restrict__ W1f,  unsigned short* __restrict__ W1f2,
    unsigned short* __restrict__ Wsim, unsigned short* __restrict__ WfA,
    unsigned short* __restrict__ WfX,  unsigned short* __restrict__ WfH,
    float* __restrict__ b1e, float* __restrict__ b1e2,
    float* __restrict__ bFX, float* __restrict__ bFH,
    float* __restrict__ Xsum)
{
    __shared__ float s1[CPAD_], t1[CPAD_], s2[CPAD_], t2[CPAD_];
    int tid = threadIdx.x;
    for (int c = tid; c < CPAD_; c += 256) {
        if (c < CI_) {
            float a1 = bn1_g[c] * rsqrtf(bn1_v[c] + 1e-5f);
            s1[c] = a1; t1[c] = bn1_b[c] - bn1_m[c] * a1;
            float a2 = bn2_g[c] * rsqrtf(bn2_v[c] + 1e-5f);
            s2[c] = a2; t2[c] = bn2_b[c] - bn2_m[c] * a2;
        } else { s1[c] = t1[c] = s2[c] = t2[c] = 0.0f; }
    }
    __syncthreads();
    for (int i = tid; i < B_ * CO_; i += 256) Xsum[i] = 0.0f;
    if (tid < CO_) {
        int o = tid;
        float a1 = b_lin[o], a2 = b_lin[o];
        for (int c = 0; c < CPAD_; ++c) {
            float w = (c < CI_) ? W_lin[o * CI_ + c] : 0.0f;
            W1f [o * CPAD_ + c] = f2bf(w * s1[c]);
            W1f2[o * CPAD_ + c] = f2bf(w * s2[c]);
            a1 += w * t1[c]; a2 += w * t2[c];
        }
        b1e[o] = a1; b1e2[o] = a2;
        for (int c = 0; c < CO_; ++c) {
            Wsim[o * CO_ + c] = f2bf(W_sim[o * CO_ + c]);
            WfA [o * CO_ + c] = f2bf(W_final[o * (CO_ + CI_) + c]);
        }
        float ax = b_final[o], ah = b_final[o];
        for (int c = 0; c < CPAD_; ++c) {
            float w = (c < CI_) ? W_final[o * (CO_ + CI_) + CO_ + c] : 0.0f;
            WfX[o * CPAD_ + c] = f2bf(w * s1[c]);
            WfH[o * CPAD_ + c] = f2bf(w * s2[c]);
            ax += w * t1[c]; ah += w * t2[c];
        }
        bFX[o] = ax; bFH[o] = ah;
    }
}

// ---------------------------------------------------------------------------
// f32 [B][Cin][N] -> bf16 [B][N][Cpad] (transposed, zero-padded channels).
// ---------------------------------------------------------------------------
__global__ __launch_bounds__(256) void transpose_pad_kernel(
    const float* __restrict__ in, unsigned short* __restrict__ out,
    int Cin, int Cpad, int N)
{
    __shared__ unsigned short tile[32][33];
    const int b  = blockIdx.z;
    const int n0 = blockIdx.x * 32;
    const int c0 = blockIdx.y * 32;
    const int col = threadIdx.x & 31;
    const int row = threadIdx.x >> 5;      // 0..7
    for (int rr = row; rr < 32; rr += 8) {
        int c = c0 + rr;
        float v = (c < Cin) ? in[((size_t)b * Cin + c) * N + n0 + col] : 0.0f;
        tile[rr][col] = f2bf(v);
    }
    __syncthreads();
    for (int rr = row; rr < 32; rr += 8) {
        int n = n0 + rr;
        out[((size_t)b * N + n) * Cpad + c0 + col] = tile[col][rr];
    }
}

// f32 [B][CO][NN] -> prelu -> bf16 [B][NN][CO] (transposed)
__global__ __launch_bounds__(256) void prelu_to_bf16_T_kernel(
    const float* __restrict__ in, const float* __restrict__ a,
    unsigned short* __restrict__ out, long long total)
{
    long long i = (long long)blockIdx.x * 256 + threadIdx.x;
    if (i >= total) return;
    int n = (int)(i % NN_);
    long long r = i / NN_;
    int o = (int)(r % CO_);
    int b = (int)(r / CO_);
    float v = in[i];
    float al = a[0];
    if (v < 0.0f) v *= al;
    out[((size_t)b * NN_ + n) * CO_ + o] = f2bf(v);
}

// ---------------------------------------------------------------------------
// WMMA bf16 GEMM:  out[b,o,n] = act( sum_k W1[o,k]B1[b,n,k] (+ W2,B2) + bias[o] )
// Double-buffered LDS, async global->LDS staging (ASYNCcnt), one barrier per
// k-tile. M = 128 fixed, block tile 128x64, 8 wave32 (16x64 strip per wave).
// ---------------------------------------------------------------------------
__global__ __launch_bounds__(256) void wmma_gemm2_kernel(
    const unsigned short* __restrict__ W1, int K1,
    const unsigned short* __restrict__ B1,      // [batch][N][K1]
    const unsigned short* __restrict__ W2, int K2,
    const unsigned short* __restrict__ B2,      // [batch][N][K2]
    const float* __restrict__ bias,
    const float* __restrict__ prelu_a,
    float* __restrict__ outF,                   // [batch][CO][N]
    unsigned short* __restrict__ outBfT,        // [batch][N][CO] (optional)
    float* __restrict__ sumOut,                 // [batch][CO]    (optional)
    int N)
{
    __shared__ __align__(16) unsigned short lA[2][128 * 32];  // 2 x 8 KB, [m][k]
    __shared__ __align__(16) unsigned short lB[2][64 * 32];   // 2 x 4 KB, [n][k]

    const int bi      = blockIdx.z;
    const int n0      = blockIdx.x * 64;
    const int tid     = threadIdx.x;
    const int wave    = tid >> 5;
    const int lane    = tid & 31;
    const int mwave   = wave * 16;
    const int halfsel = lane >> 4;
    const int mlane   = lane & 15;

    const int T1 = K1 >> 5;
    const int T2 = (W2 != nullptr) ? (K2 >> 5) : 0;
    const int T  = T1 + T2;

    const unsigned short* Bb1 = B1 + (size_t)bi * (size_t)N * (size_t)K1;
    const unsigned short* Bb2 = B2 ? B2 + (size_t)bi * (size_t)N * (size_t)K2 : nullptr;

    FragC acc[4];
    #pragma unroll
    for (int nt = 0; nt < 4; ++nt)
        #pragma unroll
        for (int r = 0; r < 8; ++r) acc[nt].f[r] = 0.0f;

    auto issue = [&](int t, int buf) {
        const unsigned short* Wp; const unsigned short* Bb; int K; int kk;
        if (t < T1) { Wp = W1; Bb = Bb1; K = K1; kk = t << 5; }
        else        { Wp = W2; Bb = Bb2; K = K2; kk = (t - T1) << 5; }
        // A (weights) tile: 128 m x 32 k = 512 x 16B; 2 chunks per thread
        const int i1 = tid + 256;
        cp16_lds(&lA[buf][(size_t)tid * 8],
                 Wp + (size_t)(tid >> 2) * K + kk + (tid & 3) * 8);
        cp16_lds(&lA[buf][(size_t)i1 * 8],
                 Wp + (size_t)(i1 >> 2) * K + kk + (i1 & 3) * 8);
        // B (activations) tile: 64 n x 32 k = 256 x 16B; 1 chunk per thread
        cp16_lds(&lB[buf][(size_t)tid * 8],
                 Bb + (size_t)(n0 + (tid >> 2)) * K + kk + (tid & 3) * 8);
    };

    auto compute = [&](int buf) {
        FragAB a;  // two contiguous 16B chunks (ISA 7.12.2 bf16 A layout)
        const unsigned short* ap = &lA[buf][(mwave + mlane) * 32 + halfsel * 8];
        a.q[0] = *(const uint4*)ap;
        a.q[1] = *(const uint4*)(ap + 16);
        #pragma unroll
        for (int nt = 0; nt < 4; ++nt) {
            FragAB b;
            const unsigned short* bp = &lB[buf][(nt * 16 + mlane) * 32 + halfsel * 8];
            b.q[0] = *(const uint4*)bp;
            b.q[1] = *(const uint4*)(bp + 16);
            acc[nt].v = __builtin_amdgcn_wmma_f32_16x16x32_bf16(
                false, a.v, false, b.v, (short)0, acc[nt].v, false, false);
        }
    };

    if (T > 0) {
        issue(0, 0);
        async_wait_all();
        __syncthreads();
        for (int t = 0; t < T; ++t) {
            const int cur = t & 1;
            if (t + 1 < T) issue(t + 1, cur ^ 1);   // overlap with compute
            compute(cur);
            async_wait_all();
            __syncthreads();
        }
    }

    const float slope = prelu_a ? prelu_a[0] : 1.0f;
    #pragma unroll
    for (int r = 0; r < 8; ++r) {
        const int o = mwave + halfsel * 8 + r;        // C/D layout, ISA 7.12.2
        const float bval = bias ? bias[o] : 0.0f;
        float rowsum = 0.0f;
        #pragma unroll
        for (int nt = 0; nt < 4; ++nt) {
            float v = acc[nt].f[r] + bval;
            if (prelu_a && v < 0.0f) v *= slope;
            const int n = n0 + nt * 16 + mlane;
            if (outF)   outF[((size_t)bi * CO_ + o) * (size_t)N + n] = v;
            if (outBfT) outBfT[((size_t)bi * N + n) * CO_ + o] = f2bf(v);
            rowsum += v;
        }
        if (sumOut) {  // fused partial reduction for X_mean
            #pragma unroll
            for (int m = 1; m < 16; m <<= 1) rowsum += __shfl_xor(rowsum, m, 32);
            if (mlane == 0) atomicAdd(&sumOut[bi * CO_ + o], rowsum);
        }
    }
}

// ---------------------------------------------------------------------------
// Attention/gating chain, one block per batch.
// ---------------------------------------------------------------------------
__global__ __launch_bounds__(1024) void attention_kernel(
    const float* __restrict__ Xsum, const float* __restrict__ Hn,
    const float* __restrict__ Hsim, const float* __restrict__ W_sim,
    const float* __restrict__ b_sim, const float* __restrict__ W_score,
    const float* __restrict__ b_score,
    float* __restrict__ Xaggre, float* __restrict__ Score)
{
    __shared__ float xmean[CO_], xsim[CO_], sim[NN_ + 1], red[1024];
    __shared__ float smax, ssum;
    const int b = blockIdx.x, tid = threadIdx.x;

    if (tid < CO_) xmean[tid] = Xsum[b * CO_ + tid] * (1.0f / HW_);
    __syncthreads();
    if (tid < CO_) {
        float a = b_sim[tid];
        for (int c = 0; c < CO_; ++c) a += W_sim[tid * CO_ + c] * xmean[c];
        xsim[tid] = a;
    }
    __syncthreads();
    {
        float a = 0.0f;
        const float* col = Hsim + (size_t)b * CO_ * NN_ + tid;
        for (int c = 0; c < CO_; ++c) a += xsim[c] * col[(size_t)c * NN_];
        sim[tid + 1] = sigmoidf_(a);
        if (tid == 0) {
            float a0 = 0.0f;
            for (int c = 0; c < CO_; ++c) a0 += xsim[c] * xsim[c];
            sim[0] = sigmoidf_(a0);
        }
    }
    __syncthreads();
    red[tid] = sim[tid];
    __syncthreads();
    if (tid == 0) red[0] = fmaxf(red[0], sim[NN_]);
    __syncthreads();
    for (int s = 512; s > 0; s >>= 1) {
        if (tid < s) red[tid] = fmaxf(red[tid], red[tid + s]);
        __syncthreads();
    }
    if (tid == 0) smax = red[0];
    __syncthreads();
    red[tid] = __expf(sim[tid] - smax);
    __syncthreads();
    if (tid == 0) red[0] += __expf(sim[NN_] - smax);
    __syncthreads();
    for (int s = 512; s > 0; s >>= 1) {
        if (tid < s) red[tid] += red[tid + s];
        __syncthreads();
    }
    if (tid == 0) ssum = red[0];
    __syncthreads();
    const float inv = 1.0f / ssum;
    sim[tid] = __expf(sim[tid] - smax) * inv;
    if (tid == 0) sim[NN_] = __expf(sim[NN_] - smax) * inv;
    __syncthreads();
    if (tid < CO_) {
        float ag = sim[0] * xmean[tid];
        const float* row = Hn + ((size_t)b * CO_ + tid) * NN_;
        for (int n = 0; n < NN_; ++n) ag += sim[n + 1] * row[n];
        xmean[tid] = ag;
        Xaggre[b * CO_ + tid] = ag;
    }
    __syncthreads();
    if (tid < CO_) {
        float a = b_score[tid];
        for (int c = 0; c < CO_; ++c) a += W_score[tid * CO_ + c] * xmean[c];
        Score[b * CO_ + tid] = sigmoidf_(a);
    }
}

// ---------------------------------------------------------------------------
// Depthwise 3x3 + gated blend + PReLU -> bf16 Xg TRANSPOSED [B][HW][CO].
// ---------------------------------------------------------------------------
__global__ __launch_bounds__(256) void dwconv_blend_kernel(
    const float* __restrict__ X, const float* __restrict__ Wdw,
    const float* __restrict__ bdw, const float* __restrict__ Score,
    const float* __restrict__ Xaggre, const float* __restrict__ act_a,
    unsigned short* __restrict__ XgT)
{
    const int b = blockIdx.z, c = blockIdx.y;
    const int p = blockIdx.x * 256 + threadIdx.x;
    const int y = p >> 7, x = p & (WW_ - 1);
    const float* Xc = X + ((size_t)b * CO_ + c) * (size_t)HW_;
    float acc = bdw[c];
    #pragma unroll
    for (int ky = 0; ky < 3; ++ky) {
        int yy = y + ky - 1;
        if ((unsigned)yy >= HH_) continue;
        #pragma unroll
        for (int kx = 0; kx < 3; ++kx) {
            int xx = x + kx - 1;
            if ((unsigned)xx >= WW_) continue;
            acc += Wdw[c * 9 + ky * 3 + kx] * Xc[yy * WW_ + xx];
        }
    }
    const float s  = Score[b * CO_ + c];
    const float ag = Xaggre[b * CO_ + c];
    float v = s * ag + (1.0f - s) * acc;
    const float al = act_a[0];
    if (v < 0.0f) v *= al;
    XgT[((size_t)b * HW_ + p) * CO_ + c] = f2bf(v);
}

// ---------------------------------------------------------------------------
extern "C" void kernel_launch(void* const* d_in, const int* in_sizes, int n_in,
                              void* d_out, int out_size, void* d_ws, size_t ws_size,
                              hipStream_t stream)
{
    (void)in_sizes; (void)n_in; (void)out_size; (void)ws_size;
    const float* X_in    = (const float*)d_in[0];
    const float* H_in    = (const float*)d_in[1];
    const float* bn1_g   = (const float*)d_in[2];
    const float* bn1_b   = (const float*)d_in[3];
    const float* bn1_m   = (const float*)d_in[4];
    const float* bn1_v   = (const float*)d_in[5];
    const float* bn2_g   = (const float*)d_in[6];
    const float* bn2_b   = (const float*)d_in[7];
    const float* bn2_m   = (const float*)d_in[8];
    const float* bn2_v   = (const float*)d_in[9];
    const float* W_lin   = (const float*)d_in[10];
    const float* b_lin   = (const float*)d_in[11];
    const float* W_sim   = (const float*)d_in[12];
    const float* b_sim   = (const float*)d_in[13];
    const float* W_score = (const float*)d_in[14];
    const float* b_score = (const float*)d_in[15];
    const float* W_dw    = (const float*)d_in[16];
    const float* b_dw    = (const float*)d_in[17];
    const float* W_final = (const float*)d_in[18];
    const float* b_final = (const float*)d_in[19];
    const float* act_a   = (const float*)d_in[20];
    const float* act2_a  = (const float*)d_in[21];
    const float* final_a = (const float*)d_in[22];

    char*  ws  = (char*)d_ws;
    size_t off = 0;
    auto alloc = [&](size_t bytes) -> char* {
        char* p = ws + off;
        off = (off + bytes + 255) & ~(size_t)255;
        return p;
    };
    unsigned short* W1f   = (unsigned short*)alloc((size_t)CO_ * CPAD_ * 2);
    unsigned short* W1f2  = (unsigned short*)alloc((size_t)CO_ * CPAD_ * 2);
    unsigned short* WsimB = (unsigned short*)alloc((size_t)CO_ * CO_ * 2);
    unsigned short* WfA   = (unsigned short*)alloc((size_t)CO_ * CO_ * 2);
    unsigned short* WfX   = (unsigned short*)alloc((size_t)CO_ * CPAD_ * 2);
    unsigned short* WfH   = (unsigned short*)alloc((size_t)CO_ * CPAD_ * 2);
    float* b1e    = (float*)alloc(CO_ * 4);
    float* b1e2   = (float*)alloc(CO_ * 4);
    float* bFX    = (float*)alloc(CO_ * 4);
    float* bFH    = (float*)alloc(CO_ * 4);
    float* Xsum   = (float*)alloc((size_t)B_ * CO_ * 4);
    float* Xaggre = (float*)alloc((size_t)B_ * CO_ * 4);
    float* Score  = (float*)alloc((size_t)B_ * CO_ * 4);
    unsigned short* XbfT  = (unsigned short*)alloc((size_t)B_ * HW_ * CPAD_ * 2);
    unsigned short* HbfT  = (unsigned short*)alloc((size_t)B_ * NN_ * CPAD_ * 2);
    float*          Xf    = (float*)alloc((size_t)B_ * CO_ * HW_ * 4);
    float*          Hn    = (float*)alloc((size_t)B_ * CO_ * NN_ * 4);
    unsigned short* HnbfT = (unsigned short*)alloc((size_t)B_ * NN_ * CO_ * 2);
    unsigned short* HnPT  = (unsigned short*)alloc((size_t)B_ * NN_ * CO_ * 2);
    float*          Hsim  = (float*)alloc((size_t)B_ * CO_ * NN_ * 4);
    unsigned short* XgT   = (unsigned short*)alloc((size_t)B_ * HW_ * CO_ * 2);

    float* outX = (float*)d_out;
    float* outH = outX + (size_t)B_ * CO_ * HW_;

    // 1) fold BN into weights, effective biases, zero X_mean accumulator
    prep_params_kernel<<<1, 256, 0, stream>>>(
        bn1_g, bn1_b, bn1_m, bn1_v, bn2_g, bn2_b, bn2_m, bn2_v,
        W_lin, b_lin, W_sim, W_final, b_final,
        W1f, W1f2, WsimB, WfA, WfX, WfH, b1e, b1e2, bFX, bFH, Xsum);

    // 2) bf16 transposed activation copies [B][N][Kpad]
    transpose_pad_kernel<<<dim3(HW_ / 32, CPAD_ / 32, B_), 256, 0, stream>>>(
        X_in, XbfT, CI_, CPAD_, HW_);
    transpose_pad_kernel<<<dim3(NN_ / 32, CPAD_ / 32, B_), 256, 0, stream>>>(
        H_in, HbfT, CI_, CPAD_, NN_);

    // 3) Stage A: X = W1f @ BN(X)  (+ fused X_mean partial sums)
    wmma_gemm2_kernel<<<dim3(HW_ / 64, 1, B_), 256, 0, stream>>>(
        W1f, CPAD_, XbfT, nullptr, 0, nullptr, b1e, nullptr,
        Xf, nullptr, Xsum, HW_);

    // 4) Hn = W1f2 @ BN(H)  (f32 [o][n] + transposed bf16 [n][o])
    wmma_gemm2_kernel<<<dim3(NN_ / 64, 1, B_), 256, 0, stream>>>(
        W1f2, CPAD_, HbfT, nullptr, 0, nullptr, b1e2, nullptr,
        Hn, HnbfT, nullptr, NN_);

    // 5) H_sim = W_sim @ Hn
    wmma_gemm2_kernel<<<dim3(NN_ / 64, 1, B_), 256, 0, stream>>>(
        WsimB, CO_, HnbfT, nullptr, 0, nullptr, b_sim, nullptr,
        Hsim, nullptr, nullptr, NN_);

    // 6) attention / softmax / aggregate / score
    attention_kernel<<<B_, 1024, 0, stream>>>(
        Xsum, Hn, Hsim, W_sim, b_sim, W_score, b_score, Xaggre, Score);

    // 7) depthwise conv + gated blend + PReLU -> XgT (bf16 transposed)
    dwconv_blend_kernel<<<dim3(HW_ / 256, CO_, B_), 256, 0, stream>>>(
        Xf, W_dw, b_dw, Score, Xaggre, act_a, XgT);

    // 8) prelu(Hn) -> bf16 transposed [n][o]
    {
        long long tot = (long long)B_ * CO_ * NN_;
        prelu_to_bf16_T_kernel<<<(unsigned)((tot + 255) / 256), 256, 0, stream>>>(
            Hn, act2_a, HnPT, tot);
    }

    // 9) X_out = prelu(WfA@Xg + WfX@BN(X) + bFX)   (concat GEMM as dual-source)
    wmma_gemm2_kernel<<<dim3(HW_ / 64, 1, B_), 256, 0, stream>>>(
        WfA, CO_, XgT, WfX, CPAD_, XbfT, bFX, final_a,
        outX, nullptr, nullptr, HW_);

    // 10) H_out = prelu(WfA@prelu(Hn) + WfH@BN(H) + bFH)
    wmma_gemm2_kernel<<<dim3(NN_ / 64, 1, B_), 256, 0, stream>>>(
        WfA, CO_, HnPT, WfH, CPAD_, HbfT, bFH, final_a,
        outH, nullptr, nullptr, NN_);
}
